// TinyMoE_75479755260558
// MI455X (gfx1250) — matrix operand order repaired and checked by reading, not compiled, not run on hardware
//
#include <hip/hip_runtime.h>
#include <hip/hip_bf16.h>

typedef __attribute__((ext_vector_type(16))) _Float16 v16h;
typedef __attribute__((ext_vector_type(8)))  _Float16 v8h;
typedef __attribute__((ext_vector_type(8)))  float    v8f;
typedef __attribute__((ext_vector_type(4)))  float    v4f;

#define EMB 16
#define NE  8
#define HID 64
#define WAVES_PER_BLOCK 4

// ---------------- LDS layout (bytes) ----------------
// W1 B-operand images: per (expert, ntile): 32 lanes x 32B  (32x16 f16 B tile, K padded 16->32)
#define OFF_W1IMG 0
#define SZ_W1IMG  (NE * 4 * 1024)          // 32768
// W2 B-operand images: per (expert, ktile): 32 lanes x 32B  (32x16 f16 B tile)
#define OFF_W2IMG (OFF_W1IMG + SZ_W1IMG)   // 32768
#define SZ_W2IMG  (NE * 2 * 1024)          // 16384
#define OFF_WR    (OFF_W2IMG + SZ_W2IMG)   // 49152 : Wr [16][8] f32
#define SZ_WR     (16 * 8 * 4)
#define OFF_BR    (OFF_WR + SZ_WR)         // 49664 : br [8] f32
#define SZ_BR     32
#define OFF_B1    (OFF_BR + SZ_BR)         // 49696 : b1 [8][64] f32
#define SZ_B1     (NE * HID * 4)
#define OFF_B2    (OFF_B1 + SZ_B1)         // 51744 : b2 [8][16] f32
#define SZ_B2     (NE * EMB * 4)
#define OFF_WT    (OFF_B2 + SZ_B2)         // 52256 : per-wave gate table [8 experts][16 tok] f32
#define SZ_WT     (WAVES_PER_BLOCK * NE * 16 * 4)
#define HROW      144                      // padded row stride for h staging (16B aligned, bank-spread)
#define OFF_HS    (OFF_WT + SZ_WT)         // 54304 : per-wave h stage [16 tok][64 hid] f16, row stride 144B
#define SZ_HS     (WAVES_PER_BLOCK * 16 * HROW)
#define SMEM_TOTAL (OFF_HS + SZ_HS)        // 63520

__global__ __launch_bounds__(WAVES_PER_BLOCK * 32)
void TinyMoE_wmma_kernel(const float* __restrict__ x,
                         const float* __restrict__ Wr,
                         const float* __restrict__ br,
                         const float* __restrict__ W1,
                         const float* __restrict__ b1,
                         const float* __restrict__ W2,
                         const float* __restrict__ b2,
                         float* __restrict__ out,
                         int numTiles)
{
    __shared__ __align__(16) unsigned char smem[SMEM_TOTAL];
    const int tid = threadIdx.x;

    // ---------------- setup: build weight operand images in LDS ----------------
    // W1 image: B tile for GEMM1 (rows K = emb 0..31 [16..31 zero], cols N = hidden chunk)
    for (int c = tid; c < NE * 4 * 32; c += WAVES_PER_BLOCK * 32) {
        const int e = c >> 7, rem = c & 127, t = rem >> 5, l = rem & 31;
        _Float16* dst = (_Float16*)(smem + OFF_W1IMG + c * 32);
        const int n = l & 15, hb = l >> 4;
        #pragma unroll
        for (int j = 0; j < 16; ++j)       // element j<8 -> K = 8*hb + j, col N = 16t + n; j>=8 -> K 16..31 zero padding
            dst[j] = (j < 8) ? (_Float16)W1[(e * 16 + 8 * hb + j) * 64 + 16 * t + n] : (_Float16)0.0f;
    }
    // W2 image: B tile for GEMM2 (rows K = hidden 32kt..32kt+31, cols N = emb)
    for (int c = tid; c < NE * 2 * 32; c += WAVES_PER_BLOCK * 32) {
        const int e = c >> 6, rem = c & 63, kt = rem >> 5, l = rem & 31;
        const int n = l & 15, hh = l >> 4;
        _Float16* dst = (_Float16*)(smem + OFF_W2IMG + c * 32);
        #pragma unroll
        for (int j = 0; j < 16; ++j)           // element j -> K_local = 8*hh + j (j<8), 16 + 8*hh + (j-8) (j>=8)
            dst[j] = (_Float16)W2[(e * 64 + 32 * kt + 8 * hh + j + (j >= 8 ? 8 : 0)) * 16 + n];
    }
    {
        float* wrS = (float*)(smem + OFF_WR);
        for (int i = tid; i < 16 * 8; i += WAVES_PER_BLOCK * 32) wrS[i] = Wr[i];
        float* brS = (float*)(smem + OFF_BR);
        if (tid < 8) brS[tid] = br[tid];
        float* b1S = (float*)(smem + OFF_B1);
        for (int i = tid; i < NE * HID; i += WAVES_PER_BLOCK * 32) b1S[i] = b1[i];
        float* b2S = (float*)(smem + OFF_B2);
        if (tid < NE * EMB) b2S[tid] = b2[tid];
    }
    __syncthreads();

    const int lane = tid & 31;
    const int wv   = tid >> 5;
    const int hh   = lane >> 4;          // lane half: selects K-offset 0/8
    const int col  = lane & 15;          // N column / token index within tile
    unsigned char* hbase = smem + OFF_HS + wv * (16 * HROW);
    float* wt = (float*)(smem + OFF_WT) + wv * (NE * 16);
    const float* wrS = (const float*)(smem + OFF_WR);
    const float* brS = (const float*)(smem + OFF_BR);
    const float* b1S = (const float*)(smem + OFF_B1);
    const float* b2S = (const float*)(smem + OFF_B2);

    const int gw = blockIdx.x * WAVES_PER_BLOCK + wv;
    const int sw = gridDim.x * WAVES_PER_BLOCK;

    for (int tile = gw; tile < numTiles; tile += sw) {
        const float* xt = x + (size_t)tile * 256;   // 16 tokens * 16 features

        // ---- load x tile: lane -> token=col, features 8*hh .. 8*hh+7 ----
        v4f xa = *(const v4f*)(xt + col * 16 + hh * 8);
        v4f xb = *(const v4f*)(xt + col * 16 + hh * 8 + 4);
        float xf[8];
        #pragma unroll
        for (int j = 0; j < 4; ++j) { xf[j] = xa[j]; xf[j + 4] = xb[j]; }

        v16h xA;                                    // A operand, K 16..31 zero
        #pragma unroll
        for (int j = 0; j < 8; ++j) { xA[j] = (_Float16)xf[j]; xA[j + 8] = (_Float16)0.0f; }

        // ---- router: partial logits over this lane's 8 features ----
        float pl[8] = {0.f, 0.f, 0.f, 0.f, 0.f, 0.f, 0.f, 0.f};
        #pragma unroll
        for (int k = 0; k < 8; ++k) {
            const float* row = wrS + (hh * 8 + k) * 8;
            v4f w0 = *(const v4f*)row;
            v4f w1 = *(const v4f*)(row + 4);
            #pragma unroll
            for (int e = 0; e < 4; ++e) { pl[e] += xf[k] * w0[e]; pl[e + 4] += xf[k] * w1[e]; }
        }
        float lg[8];
        #pragma unroll
        for (int e = 0; e < 8; ++e)
            lg[e] = pl[e] + __shfl_xor(pl[e], 16, 32) + brS[e];   // combine lane halves

        float mx = lg[0];
        #pragma unroll
        for (int e = 1; e < 8; ++e) mx = fmaxf(mx, lg[e]);
        float ex[8], sum = 0.f;
        #pragma unroll
        for (int e = 0; e < 8; ++e) { ex[e] = __expf(lg[e] - mx); sum += ex[e]; }
        const float inv = 1.0f / sum;

        int i1 = 0; float v1 = ex[0];
        #pragma unroll
        for (int e = 1; e < 8; ++e) if (ex[e] > v1) { v1 = ex[e]; i1 = e; }
        int i2 = (i1 == 0) ? 1 : 0; float v2 = ex[i2];
        #pragma unroll
        for (int e = 0; e < 8; ++e) if (e != i1 && ex[e] > v2) { v2 = ex[e]; i2 = e; }

        if (lane < 16) {
            #pragma unroll
            for (int e = 0; e < 8; ++e)
                wt[e * 16 + col] = (e == i1 || e == i2) ? ex[e] * inv : 0.0f;
        }
        asm volatile("" ::: "memory");   // keep LDS store->load order (HW is in-order per wave)

        v8f acc;
        #pragma unroll
        for (int r = 0; r < 8; ++r) acc[r] = 0.f;

        for (int e = 0; e < NE; ++e) {
            const float* wrow = wt + e * 16 + hh * 8;
            v4f w0 = *(const v4f*)wrow;
            v4f w1 = *(const v4f*)(wrow + 4);
            int anyw = (w0[0] != 0.f) | (w0[1] != 0.f) | (w0[2] != 0.f) | (w0[3] != 0.f) |
                       (w1[0] != 0.f) | (w1[1] != 0.f) | (w1[2] != 0.f) | (w1[3] != 0.f);
            if (__ballot(anyw)) {        // wave-uniform: skip unselected experts
                // ---- GEMM1: h = relu(x @ W1[e] + b1[e]) ----
                #pragma unroll
                for (int nt = 0; nt < 4; ++nt) {
                    const unsigned char* bi = smem + OFF_W1IMG + (e * 4 + nt) * 1024 + lane * 32;
                    v8h blo = *(const v8h*)bi;
                    v8h bhi = *(const v8h*)(bi + 16);
                    v16h bv;
                    #pragma unroll
                    for (int j = 0; j < 8; ++j) { bv[j] = blo[j]; bv[j + 8] = bhi[j]; }
                    const float b1v = b1S[e * 64 + nt * 16 + col];
                    v8f c;
                    #pragma unroll
                    for (int r = 0; r < 8; ++r) c[r] = b1v;      // bias broadcast over rows
                    v8f h = __builtin_amdgcn_wmma_f32_16x16x32_f16(
                        false, xA, false, bv, (short)0, c, false, false);
                    // relu + f16 convert + transpose-stage to LDS [token][hidden]
                    #pragma unroll
                    for (int r = 0; r < 8; ++r) {
                        float hv = fmaxf(h[r], 0.0f);
                        *(_Float16*)(hbase + (r + 8 * hh) * HROW + (nt * 16 + col) * 2) =
                            (_Float16)hv;
                    }
                }
                asm volatile("" ::: "memory");

                // ---- GEMM2: eo = h @ W2[e] ----
                v8f eo;
                #pragma unroll
                for (int r = 0; r < 8; ++r) eo[r] = 0.f;
                #pragma unroll
                for (int kt = 0; kt < 2; ++kt) {
                    const unsigned char* ap = hbase + col * HROW + kt * 64 + hh * 16;
                    v8h alo = *(const v8h*)ap;          // K_local 0..7  (+8*hh)
                    v8h ahi = *(const v8h*)(ap + 32);   // K_local 16..23 (+8*hh)
                    v16h av;
                    #pragma unroll
                    for (int j = 0; j < 8; ++j) { av[j] = alo[j]; av[j + 8] = ahi[j]; }
                    const unsigned char* bi = smem + OFF_W2IMG + (e * 2 + kt) * 1024 + lane * 32;
                    v8h blo = *(const v8h*)bi;
                    v8h bhi = *(const v8h*)(bi + 16);
                    v16h bv;
                    #pragma unroll
                    for (int j = 0; j < 8; ++j) { bv[j] = blo[j]; bv[j + 8] = bhi[j]; }
                    eo = __builtin_amdgcn_wmma_f32_16x16x32_f16(
                        false, av, false, bv, (short)0, eo, false, false);
                }

                // ---- out += (eo + b2) * gate ----
                const float b2v = b2S[e * 16 + col];
                #pragma unroll
                for (int r = 0; r < 8; ++r) {
                    const float gr = (r < 4) ? w0[r] : w1[r - 4];
                    acc[r] += (eo[r] + b2v) * gr;
                }
                asm volatile("" ::: "memory");   // hStage reused next expert
            }
        }

        // ---- store output tile ----
        float* op = out + (size_t)tile * 256;
        #pragma unroll
        for (int r = 0; r < 8; ++r)
            op[(r + 8 * hh) * 16 + col] = acc[r];
    }
}

extern "C" void kernel_launch(void* const* d_in, const int* in_sizes, int n_in,
                              void* d_out, int out_size, void* d_ws, size_t ws_size,
                              hipStream_t stream) {
    (void)n_in; (void)d_ws; (void)ws_size; (void)out_size;
    const float* x  = (const float*)d_in[0];
    const float* Wr = (const float*)d_in[1];
    const float* br = (const float*)d_in[2];
    const float* W1 = (const float*)d_in[3];
    const float* b1 = (const float*)d_in[4];
    const float* W2 = (const float*)d_in[5];
    const float* b2 = (const float*)d_in[6];
    float* out = (float*)d_out;

    const int tokens   = in_sizes[0] / EMB;       // 262144
    const int numTiles = tokens / 16;             // 16384
    const int blocks   = 1024;                    // 4096 waves, 4 tiles each
    TinyMoE_wmma_kernel<<<blocks, WAVES_PER_BLOCK * 32, 0, stream>>>(
        x, Wr, br, W1, b1, W2, b2, out, numTiles);
}
